// SAGE_35150012350591
// MI455X (gfx1250) — compile-verified, matrix operand order checked by reading, not run on hardware
//
#include <hip/hip_runtime.h>
#include <hip/hip_bf16.h>

// ---------------------------------------------------------------------------
// CDNA5 (gfx1250) GraphSAGE pipeline:
//   FF(128->256->128, BN+ReLU) -> SAGE(128->256, BN+LeakyReLU)
//   -> SAGE(256->128, BN+LeakyReLU)
// GEMMs: v_wmma_f32_16x16x32_bf16 (wave32 WMMA), fp32 accumulate, with
// double-buffered GLOBAL_LOAD_ASYNC_TO_LDS_B128 tile staging (ASYNCcnt).
// ---------------------------------------------------------------------------

typedef __attribute__((ext_vector_type(16))) __bf16 v16bf;
typedef __attribute__((ext_vector_type(8)))  float  v8f;
typedef __attribute__((address_space(3))) unsigned short lds_u16;

#define N_NODES 50000
#define N_EDGES 600000

__device__ __forceinline__ unsigned short f32_to_bf16_rne(float f) {
    unsigned int u = __float_as_uint(f);
    unsigned int lsb = (u >> 16) & 1u;
    u += 0x7fffu + lsb;                 // round to nearest even
    return (unsigned short)(u >> 16);
}

// Async global -> LDS copy of 16 bytes (per lane). LDS dest given as byte
// offset (VDST vgpr), global source as 64-bit address. Tracked by ASYNCcnt.
__device__ __forceinline__ void async_copy16(unsigned lds_off,
                                             const unsigned short* g) {
    asm volatile("global_load_async_to_lds_b128 %0, %1, off"
                 :: "v"(lds_off), "v"(g)
                 : "memory");
}
__device__ __forceinline__ void wait_async0() {
    asm volatile("s_wait_asynccnt 0x0" ::: "memory");
}

// ---------------------------------------------------------------------------
// WMMA GEMM: C[M,N] = A[M,K](bf16, row-major) x B(bf16, packed col-major:
// Bp[n*K+k]) + bias[N].  Block tile 128x64, 8 waves, wave tile 32x32.
// Double-buffered async LDS staging.
// ---------------------------------------------------------------------------
#define BM 128
#define BN 64
#define BK 32
#define LDST 40   // padded LDS row stride in bf16 elements (80 B)

__global__ __launch_bounds__(256)
void gemm_bf16_wmma(const unsigned short* __restrict__ A,
                    const unsigned short* __restrict__ Bp,
                    const float* __restrict__ bias,
                    float* __restrict__ C,
                    int M, int N, int K)
{
    __shared__ unsigned short lA[2][BM * LDST];
    __shared__ unsigned short lB[2][BN * LDST];

    const int tid  = threadIdx.x;
    const int wave = tid >> 5;
    const int lane = tid & 31;
    const int wy   = wave >> 1;       // 0..3 : 32-row strip
    const int wx   = wave & 1;        // 0..1 : 32-col strip
    const int rowBase = blockIdx.y * BM;
    const int colBase = blockIdx.x * BN;

    // LDS byte offsets of the two staging buffers (for async VDST operand)
    const unsigned baseA = (unsigned)(size_t)(lds_u16*)&lA[0][0];
    const unsigned baseB = (unsigned)(size_t)(lds_u16*)&lB[0][0];
    const unsigned bufSzA = (unsigned)(BM * LDST * 2);
    const unsigned bufSzB = (unsigned)(BN * LDST * 2);

    // Per-thread staging assignments (row-clamped => no exec-mask branching;
    // clamped rows only feed output rows the epilogue never stores).
    const int rA0 = tid >> 2;                // A chunk 0 row
    const int pA0 = tid & 3;                 // A chunk 0 16B-part
    const int rA1 = (tid + 256) >> 2;        // A chunk 1 row
    const int pA1 = pA0;
    const int gA0 = min(rowBase + rA0, M - 1);
    const int gA1 = min(rowBase + rA1, M - 1);
    const int nB  = tid >> 2;                // B col (colBase+nB always < N)
    const int pB  = tid & 3;
    const int gB  = colBase + nB;

    auto stage = [&](int buf, int k0) {
        async_copy16(baseA + (unsigned)buf * bufSzA +
                         (unsigned)(rA0 * LDST * 2 + pA0 * 16),
                     A + (size_t)gA0 * K + k0 + pA0 * 8);
        async_copy16(baseA + (unsigned)buf * bufSzA +
                         (unsigned)(rA1 * LDST * 2 + pA1 * 16),
                     A + (size_t)gA1 * K + k0 + pA1 * 8);
        async_copy16(baseB + (unsigned)buf * bufSzB +
                         (unsigned)(nB * LDST * 2 + pB * 16),
                     Bp + (size_t)gB * K + k0 + pB * 8);
    };

    const v8f vzero = {0.f, 0.f, 0.f, 0.f, 0.f, 0.f, 0.f, 0.f};
    v8f acc[2][2];
    acc[0][0] = vzero; acc[0][1] = vzero; acc[1][0] = vzero; acc[1][1] = vzero;

    const int half = lane >> 4;       // K-half select per ISA 16-bit layout
    const int l15  = lane & 15;

    // Prologue: stage tile 0 into buffer 0
    stage(0, 0);
    wait_async0();
    __syncthreads();

    int buf = 0;
    for (int k0 = 0; k0 < K; k0 += BK) {
        // Prefetch next K-tile into the other buffer (overlaps with compute)
        if (k0 + BK < K) stage(buf ^ 1, k0 + BK);

        #pragma unroll
        for (int i = 0; i < 2; ++i) {
            // A fragment: lane l<16 holds row m, K = {kh*8..+7, 16+kh*8..+7}
            int m = wy * 32 + i * 16 + l15;
            const unsigned short* pa = &lA[buf][m * LDST + half * 8];
            v16bf afrag;
            ((uint4*)&afrag)[0] = *(const uint4*)(pa);
            ((uint4*)&afrag)[1] = *(const uint4*)(pa + 16);
            #pragma unroll
            for (int j = 0; j < 2; ++j) {
                int n = wx * 32 + j * 16 + l15;
                const unsigned short* pb = &lB[buf][n * LDST + half * 8];
                v16bf bfrag;
                ((uint4*)&bfrag)[0] = *(const uint4*)(pb);
                ((uint4*)&bfrag)[1] = *(const uint4*)(pb + 16);
                acc[i][j] = __builtin_amdgcn_wmma_f32_16x16x32_bf16(
                    false, afrag, false, bfrag, (short)0, acc[i][j],
                    false, false);
            }
        }

        // My async writes to buf^1 done; barrier makes all waves' visible.
        wait_async0();
        __syncthreads();
        buf ^= 1;
    }

    // Epilogue: C/D layout = VGPR r -> (M = r + 8*half), N = l15
    #pragma unroll
    for (int i = 0; i < 2; ++i) {
        #pragma unroll
        for (int j = 0; j < 2; ++j) {
            int col = colBase + wx * 32 + j * 16 + l15;
            float bv = bias[col];
            #pragma unroll
            for (int r = 0; r < 8; ++r) {
                int row = rowBase + wy * 32 + i * 16 + half * 8 + r;
                if (row < M)
                    C[(size_t)row * N + col] = acc[i][j][r] + bv;
            }
        }
    }
}

// ---------------------------------------------------------------------------
// Weight pack: fp32 [Kt+Kb, N] row-major (two stacked matrices) ->
//              bf16 col-major  out[n*(Kt+Kb) + k]
// ---------------------------------------------------------------------------
__global__ void pack_w_bf16(const float* __restrict__ Wt,
                            const float* __restrict__ Wb,
                            unsigned short* __restrict__ out,
                            int Kt, int Kb, int N)
{
    int Ktot = Kt + Kb;
    int idx = blockIdx.x * blockDim.x + threadIdx.x;
    if (idx >= Ktot * N) return;
    int n = idx / Ktot;
    int k = idx - n * Ktot;
    float v = (k < Kt) ? Wt[(size_t)k * N + n] : Wb[(size_t)(k - Kt) * N + n];
    out[(size_t)n * Ktot + k] = f32_to_bf16_rne(v);
}

// fp32 [M,D] -> bf16 with output row stride / column offset
__global__ void to_bf16_strided(const float* __restrict__ X,
                                unsigned short* __restrict__ out,
                                int M, int D, int strideOut, int colOff)
{
    size_t idx = (size_t)blockIdx.x * blockDim.x + threadIdx.x;
    if (idx >= (size_t)M * D) return;
    int r = (int)(idx / D);
    int c = (int)(idx - (size_t)r * D);
    out[(size_t)r * strideOut + colOff + c] = f32_to_bf16_rne(X[idx]);
}

// ---------------------------------------------------------------------------
// BatchNorm column statistics (training mode, biased variance)
// ---------------------------------------------------------------------------
__global__ void col_stats(const float* __restrict__ X,
                          float* __restrict__ sum, float* __restrict__ sumsq,
                          int M, int N, int rowsPerBlock)
{
    int col = blockIdx.x * blockDim.x + threadIdx.x;
    if (col >= N) return;
    int r0 = blockIdx.y * rowsPerBlock;
    int r1 = min(M, r0 + rowsPerBlock);
    float s = 0.f, ss = 0.f;
    for (int r = r0; r < r1; ++r) {
        float v = X[(size_t)r * N + col];
        s += v;
        ss += v * v;
    }
    atomicAdd(&sum[col], s);
    atomicAdd(&sumsq[col], ss);
}

__global__ void bn_finalize(const float* __restrict__ sum,
                            const float* __restrict__ sumsq,
                            const float* __restrict__ gamma,
                            const float* __restrict__ beta,
                            float* __restrict__ scale,
                            float* __restrict__ shift,
                            int N, float invM)
{
    int c = blockIdx.x * blockDim.x + threadIdx.x;
    if (c >= N) return;
    float mu  = sum[c] * invM;
    float var = fmaxf(sumsq[c] * invM - mu * mu, 0.0f);
    float sc  = gamma[c] * rsqrtf(var + 1e-5f);
    scale[c] = sc;
    shift[c] = beta[c] - mu * sc;
}

// BN + ReLU/LeakyReLU; writes bf16 (strided concat buffer) and/or fp32
__global__ void bn_act(const float* __restrict__ X,
                       const float* __restrict__ scale,
                       const float* __restrict__ shift,
                       int M, int N, float slope,
                       unsigned short* __restrict__ dstBf, int strideBf, int colOff,
                       float* __restrict__ dstF)
{
    size_t idx = (size_t)blockIdx.x * blockDim.x + threadIdx.x;
    if (idx >= (size_t)M * N) return;
    int r = (int)(idx / N);
    int c = (int)(idx - (size_t)r * N);
    float v = X[idx] * scale[c] + shift[c];
    v = (v > 0.f) ? v : v * slope;
    if (dstBf) dstBf[(size_t)r * strideBf + colOff + c] = f32_to_bf16_rne(v);
    if (dstF)  dstF[idx] = v;
}

// ---------------------------------------------------------------------------
// Graph aggregation (mean over in-edges)
// ---------------------------------------------------------------------------
__global__ void compute_deg(const int* __restrict__ dst,
                            float* __restrict__ deg, int nE)
{
    int e = blockIdx.x * blockDim.x + threadIdx.x;
    if (e < nE) atomicAdd(&deg[dst[e]], 1.0f);
}

__global__ void edge_agg(const float* __restrict__ feat,
                         const int* __restrict__ src,
                         const int* __restrict__ dst,
                         float* __restrict__ agg, int nE, int D)
{
    size_t idx = (size_t)blockIdx.x * blockDim.x + threadIdx.x;
    int chunks = D >> 2;
    if (idx >= (size_t)nE * chunks) return;
    int e = (int)(idx / chunks);
    int c = (int)(idx - (size_t)e * chunks) * 4;
    int s = src[e];
    int d = dst[e];
    float4 v = *(const float4*)(feat + (size_t)s * D + c);
    float* p = agg + (size_t)d * D + c;
    atomicAdd(p + 0, v.x);
    atomicAdd(p + 1, v.y);
    atomicAdd(p + 2, v.z);
    atomicAdd(p + 3, v.w);
}

__global__ void neigh_fin(const float* __restrict__ agg,
                          const float* __restrict__ deg,
                          unsigned short* __restrict__ dstBf,
                          int strideBf, int colOff, int M, int D)
{
    size_t idx = (size_t)blockIdx.x * blockDim.x + threadIdx.x;
    if (idx >= (size_t)M * D) return;
    int r = (int)(idx / D);
    int c = (int)(idx - (size_t)r * D);
    float v = agg[idx] / fmaxf(deg[r], 1.0f);
    dstBf[(size_t)r * strideBf + colOff + c] = f32_to_bf16_rne(v);
}

// ---------------------------------------------------------------------------
// Host launcher
// ---------------------------------------------------------------------------
static inline size_t alignUp(size_t x) { return (x + 255) & ~(size_t)255; }

extern "C" void kernel_launch(void* const* d_in, const int* in_sizes, int n_in,
                              void* d_out, int out_size, void* d_ws, size_t ws_size,
                              hipStream_t stream)
{
    const int M = N_NODES, nE = N_EDGES;
    const float invM = 1.0f / (float)M;

    const float* nodes = (const float*)d_in[0];
    const int*   src   = (const int*)d_in[1];
    const int*   dst   = (const int*)d_in[2];
    const float* Wf1 = (const float*)d_in[3];  const float* bf1 = (const float*)d_in[4];
    const float* gf1 = (const float*)d_in[5];  const float* bef1= (const float*)d_in[6];
    const float* Wf2 = (const float*)d_in[7];  const float* bf2 = (const float*)d_in[8];
    const float* gf2 = (const float*)d_in[9];  const float* bef2= (const float*)d_in[10];
    const float* Ws1 = (const float*)d_in[11]; const float* Wn1 = (const float*)d_in[12];
    const float* b1  = (const float*)d_in[13]; const float* gs1 = (const float*)d_in[14];
    const float* bs1 = (const float*)d_in[15];
    const float* Ws2 = (const float*)d_in[16]; const float* Wn2 = (const float*)d_in[17];
    const float* b2  = (const float*)d_in[18]; const float* gs2 = (const float*)d_in[19];
    const float* bs2 = (const float*)d_in[20];
    float* out = (float*)d_out;

    // ---- workspace carve ----
    char* ws = (char*)d_ws;
    auto take = [&](size_t bytes) -> void* { void* p = ws; ws += alignUp(bytes); return p; };
    float*          bufC   = (float*)         take((size_t)M * 256 * 4); // GEMM outputs
    float*          bufF   = (float*)         take((size_t)M * 256 * 4); // fp32 activations
    float*          bufAgg = (float*)         take((size_t)M * 256 * 4); // scatter accumulator
    unsigned short* bufX   = (unsigned short*)take((size_t)M * 512 * 2); // bf16 operand A (ping)
    unsigned short* bufY   = (unsigned short*)take((size_t)M * 512 * 2); // bf16 operand A (pong)
    float*          deg    = (float*)         take((size_t)M * 4);
    float*          sumv   = (float*)         take(512 * 4);
    float*          sumsq  = (float*)         take(512 * 4);
    float*          scale  = (float*)         take(512 * 4);
    float*          shift  = (float*)         take(512 * 4);
    unsigned short* Wf1p   = (unsigned short*)take(128 * 256 * 2);
    unsigned short* Wf2p   = (unsigned short*)take(256 * 128 * 2);
    unsigned short* W1c    = (unsigned short*)take(256 * 256 * 2);
    unsigned short* W2c    = (unsigned short*)take(512 * 128 * 2);
    if ((size_t)(ws - (char*)d_ws) > ws_size) return;

    const int TB = 256;
    auto gemmGrid = [&](int N) { return dim3((unsigned)(N / BN), (unsigned)((M + BM - 1) / BM)); };
    auto elemGrid = [&](size_t n) { return dim3((unsigned)((n + TB - 1) / TB)); };

    // ---- weight packing (bf16, col-major; SAGE weights K-concatenated) ----
    pack_w_bf16<<<elemGrid(128 * 256), TB, 0, stream>>>(Wf1, nullptr, Wf1p, 128, 0, 256);
    pack_w_bf16<<<elemGrid(256 * 128), TB, 0, stream>>>(Wf2, nullptr, Wf2p, 256, 0, 128);
    pack_w_bf16<<<elemGrid(256 * 256), TB, 0, stream>>>(Ws1, Wn1, W1c, 128, 128, 256);
    pack_w_bf16<<<elemGrid(512 * 128), TB, 0, stream>>>(Ws2, Wn2, W2c, 256, 256, 128);

    // ---- degrees (shared by both SAGE layers) ----
    hipMemsetAsync(deg, 0, (size_t)M * 4, stream);
    compute_deg<<<elemGrid(nE), TB, 0, stream>>>(dst, deg, nE);

    // ---- nodes -> bf16 ----
    to_bf16_strided<<<elemGrid((size_t)M * 128), TB, 0, stream>>>(nodes, bufX, M, 128, 128, 0);

    dim3 statGrid1(1, (M + 2047) / 2048);

    // ======== FF layer 1: Y = nodes @ Wf1 + bf1 ; BN ; ReLU ========
    gemm_bf16_wmma<<<gemmGrid(256), TB, 0, stream>>>(bufX, Wf1p, bf1, bufC, M, 256, 128);
    hipMemsetAsync(sumv, 0, 512 * 4, stream);
    hipMemsetAsync(sumsq, 0, 512 * 4, stream);
    col_stats<<<statGrid1, TB, 0, stream>>>(bufC, sumv, sumsq, M, 256, 2048);
    bn_finalize<<<1, 256, 0, stream>>>(sumv, sumsq, gf1, bef1, scale, shift, 256, invM);
    bn_act<<<elemGrid((size_t)M * 256), TB, 0, stream>>>(bufC, scale, shift, M, 256, 0.0f,
                                                         bufY, 256, 0, nullptr);

    // ======== FF layer 2: Y = h @ Wf2 + bf2 ; BN ; ReLU ========
    gemm_bf16_wmma<<<gemmGrid(128), TB, 0, stream>>>(bufY, Wf2p, bf2, bufC, M, 128, 256);
    hipMemsetAsync(sumv, 0, 512 * 4, stream);
    hipMemsetAsync(sumsq, 0, 512 * 4, stream);
    col_stats<<<statGrid1, TB, 0, stream>>>(bufC, sumv, sumsq, M, 128, 2048);
    bn_finalize<<<1, 256, 0, stream>>>(sumv, sumsq, gf2, bef2, scale, shift, 128, invM);
    // h: fp32 -> bufF (for aggregation), bf16 -> bufX cols [0,128) (stride 256)
    bn_act<<<elemGrid((size_t)M * 128), TB, 0, stream>>>(bufC, scale, shift, M, 128, 0.0f,
                                                         bufX, 256, 0, bufF);

    // ======== SAGE 1 aggregation: mean_{src->dst}(h) -> bufX cols [128,256) ========
    hipMemsetAsync(bufAgg, 0, (size_t)M * 128 * 4, stream);
    edge_agg<<<elemGrid((size_t)nE * 32), TB, 0, stream>>>(bufF, src, dst, bufAgg, nE, 128);
    neigh_fin<<<elemGrid((size_t)M * 128), TB, 0, stream>>>(bufAgg, deg, bufX, 256, 128, M, 128);

    // ======== SAGE 1 GEMM: [h | h_n] @ [Ws1;Wn1] + b1 ; BN ; LeakyReLU ========
    gemm_bf16_wmma<<<gemmGrid(256), TB, 0, stream>>>(bufX, W1c, b1, bufC, M, 256, 256);
    hipMemsetAsync(sumv, 0, 512 * 4, stream);
    hipMemsetAsync(sumsq, 0, 512 * 4, stream);
    col_stats<<<statGrid1, TB, 0, stream>>>(bufC, sumv, sumsq, M, 256, 2048);
    bn_finalize<<<1, 256, 0, stream>>>(sumv, sumsq, gs1, bs1, scale, shift, 256, invM);
    // h1: fp32 -> bufF, bf16 -> bufY cols [0,256) (stride 512)
    bn_act<<<elemGrid((size_t)M * 256), TB, 0, stream>>>(bufC, scale, shift, M, 256, 0.01f,
                                                         bufY, 512, 0, bufF);

    // ======== SAGE 2 aggregation: mean(h1) -> bufY cols [256,512) ========
    hipMemsetAsync(bufAgg, 0, (size_t)M * 256 * 4, stream);
    edge_agg<<<elemGrid((size_t)nE * 64), TB, 0, stream>>>(bufF, src, dst, bufAgg, nE, 256);
    neigh_fin<<<elemGrid((size_t)M * 256), TB, 0, stream>>>(bufAgg, deg, bufY, 512, 256, M, 256);

    // ======== SAGE 2 GEMM: [h1 | h1_n] @ [Ws2;Wn2] + b2 ; BN ; LeakyReLU -> out ========
    gemm_bf16_wmma<<<gemmGrid(128), TB, 0, stream>>>(bufY, W2c, b2, bufC, M, 128, 512);
    hipMemsetAsync(sumv, 0, 512 * 4, stream);
    hipMemsetAsync(sumsq, 0, 512 * 4, stream);
    col_stats<<<statGrid1, TB, 0, stream>>>(bufC, sumv, sumsq, M, 128, 2048);
    bn_finalize<<<1, 256, 0, stream>>>(sumv, sumsq, gs2, bs2, scale, shift, 128, invM);
    bn_act<<<elemGrid((size_t)M * 128), TB, 0, stream>>>(bufC, scale, shift, M, 128, 0.01f,
                                                         nullptr, 0, 0, out);
}
